// LIIF_58703613002182
// MI455X (gfx1250) — compile-verified, matrix operand order checked by reading, not run on hardware
//
#include <hip/hip_runtime.h>
#include <hip/hip_bf16.h>
#include <hip/hip_fp16.h>

// ---------------------------------------------------------------------------
// LIIF + hash-grid MLP, fused for MI455X (gfx1250).
// 464 GFLOP of GEMM vs ~10MB of irreducible HBM traffic -> compute bound.
// Strategy: f16 WMMA (v_wmma_f32_16x16x32_f16), whole MLP fused per tile,
// weights + ping-pong activations resident in 320KB LDS.
// NEW this round: weight staging and q_feat gather use the CDNA5 async
// LDS-DMA path (global_load_async_to_lds_b128 + s_wait_asynccnt).
// ---------------------------------------------------------------------------

#define EPSF 1e-6f
#define XS   264      // LDS row stride (f16 elems) for activations (pad vs 256)
#define WSR  264      // LDS row stride for weights
#define MROWS 128     // rows per workgroup = 32 points * 4 shifts
#define PTS_PER_WG 32

typedef __attribute__((ext_vector_type(16))) _Float16 v16h;
typedef __attribute__((ext_vector_type(8)))  float    v8f;

union Frag16 { v16h v; uint4 q[2]; };

// Low 32 bits of a generic pointer into LDS == wave-relative LDS byte address
// (ISA 10.2: the LDS aperture offset is the low 32 address bits).
__device__ __forceinline__ unsigned lds_addr32(const void* p) {
  return (unsigned)(uintptr_t)p;
}

// Async DMA: LDS[lds] = MEM[saddr + voff] (16B), tracked by ASYNCcnt. GVS form.
__device__ __forceinline__ void async_lds_b128_gvs(unsigned lds, unsigned voff,
                                                   const void* sbase) {
  asm volatile("global_load_async_to_lds_b128 %0, %1, %2"
               :: "v"(lds), "v"(voff), "s"(sbase) : "memory");
}

// Async DMA: LDS[lds] = MEM[vaddr64] (16B). GV form for divergent addresses.
__device__ __forceinline__ void async_lds_b128_gv(unsigned lds,
                                                  unsigned long long vaddr) {
  asm volatile("global_load_async_to_lds_b128 %0, %1, off"
               :: "v"(lds), "v"(vaddr) : "memory");
}

__device__ __forceinline__ void wait_asynccnt0() {
  asm volatile("s_wait_asynccnt 0" ::: "memory");
}

// A-fragment (16x32 f16, M=lane&15) and B-fragment (32x16, N=lane&15) share
// the same per-lane picture when B is stored transposed [n][k]:
// lanes 0-15: K chunks {0..7},{16..23}; lanes 16-31: {8..15},{24..31}.
__device__ __forceinline__ v16h ld_frag(const _Float16* base, int stride,
                                        int r16, int k0, int lane) {
  const int rr = r16 + (lane & 15);
  const int hi = (lane >> 4) & 1;
  const _Float16* p = base + rr * stride + k0 + hi * 8;
  Frag16 f;
  f.q[0] = *(const uint4*)(p);
  f.q[1] = *(const uint4*)(p + 16);
  return f.v;
}

__device__ __forceinline__ float gelu_exact(float x) {
  return 0.5f * x * (1.0f + erff(x * 0.70710678118654752440f));
}

__device__ __forceinline__ float lin64(int k) {
  return -1.0f + EPSF + (float)k * ((2.0f - 2.0f * EPSF) / 63.0f);
}

__device__ __forceinline__ int nearest_idx(float g) {   // n = 64
  float x = (g + 1.0f) * 32.0f - 0.5f;
  int i = (int)rintf(x);                                // round half-to-even
  return min(max(i, 0), 63);
}

__device__ __forceinline__ unsigned hash2(unsigned a, unsigned b) {
  return ((a * 73856093u) ^ (b * 19349663u)) & 32767u;
}

// ------------------------- prep kernels -----------------------------------

// feat (B,C,H,W) f32 -> featT (B,H,W,C) f16
__global__ void prep_feat(const float* __restrict__ feat,
                          _Float16* __restrict__ featT) {
  const int idx = blockIdx.x * 256 + threadIdx.x;   // < 4*64*64*64
  const int x = idx & 63, y = (idx >> 6) & 63, c = (idx >> 12) & 63, b = idx >> 18;
  featT[(((long)b * 64 + y) * 64 + x) * 64 + c] = (_Float16)feat[idx];
}

// W0 (92,256)->W0t f16 [256][96] (zero-padded K), W1..3 (256,256)->Wt f16 [l][256][256]
// transposed [n][k]; W4 (256,3)->W4t f32 [3][256].
__global__ void prep_weights(const float* __restrict__ W0, const float* __restrict__ W1,
                             const float* __restrict__ W2, const float* __restrict__ W3,
                             const float* __restrict__ W4,
                             _Float16* __restrict__ W0t, _Float16* __restrict__ Wt,
                             float* __restrict__ W4t) {
  const int i = blockIdx.x * 256 + threadIdx.x;
  if (i < 24576) {                       // 256*96
    const int n = i / 96, k = i % 96;
    W0t[i] = (_Float16)((k < 92) ? W0[k * 256 + n] : 0.0f);
  } else if (i < 221184) {               // + 3*65536
    const int j = i - 24576;
    const int l = j >> 16, r = j & 65535;
    const int n = r >> 8, k = r & 255;
    const float* W = (l == 0) ? W1 : ((l == 1) ? W2 : W3);
    Wt[j] = (_Float16)W[k * 256 + n];
  } else if (i < 221952) {               // + 3*256
    const int j = i - 221184;
    const int o = j / 256, k = j % 256;
    W4t[j] = W4[k * 3 + o];
  }
}

// ------------------------- fused main kernel -------------------------------

__global__ __launch_bounds__(256)
void liif_fused(const _Float16* __restrict__ featT,
                const float* __restrict__ coord,
                const float* __restrict__ cell,
                const float* __restrict__ tables,
                const _Float16* __restrict__ W0t,
                const _Float16* __restrict__ Wt,    // [3][256][256] f16 [n][k]
                const float* __restrict__ W4t,      // [3][256] f32
                const float* __restrict__ b0, const float* __restrict__ b1,
                const float* __restrict__ b2, const float* __restrict__ b3,
                const float* __restrict__ b4,
                float* __restrict__ out) {
  __shared__ __align__(16) _Float16 Xsh[MROWS * XS];   // 67,584 B
  __shared__ __align__(16) _Float16 Ysh[MROWS * XS];   // 67,584 B
  __shared__ __align__(16) _Float16 Wsh[256 * WSR];    // 135,168 B
  __shared__ float biasSh[256];
  __shared__ float predSh[MROWS * 3];
  __shared__ float areaSh[MROWS];

  const int t = threadIdx.x;

  // -------- stage 0: build 128 input rows (thread pair per row) ----------
  {
    const int row = t >> 1;                 // 0..127
    const int p   = row & (PTS_PER_WG - 1); // local point
    const int s   = row >> 5;               // shift: (vx,vy) in {-1,1}x{-1,1}
    const long g  = (long)blockIdx.x * PTS_PER_WG + p;   // global point in B*N
    const float cy = coord[g * 2 + 0];
    const float cx = coord[g * 2 + 1];
    _Float16* dst = Xsh + row * XS;
    if ((t & 1) == 0) {
      // shift-dependent: q_feat gather (async DMA to LDS), rel_coord, area
      const float vx = (s & 2) ? 1.0f : -1.0f;
      const float vy = (s & 1) ? 1.0f : -1.0f;
      float c0 = cy + vx * (1.0f / 64.0f) + EPSF;
      float c1 = cx + vy * (1.0f / 64.0f) + EPSF;
      c0 = fminf(fmaxf(c0, -1.0f + EPSF), 1.0f - EPSF);
      c1 = fminf(fmaxf(c1, -1.0f + EPSF), 1.0f - EPSF);
      const int iy = nearest_idx(c0);
      const int ix = nearest_idx(c1);
      const int b = (int)(g >> 16);
      const _Float16* src = featT + (((long)b * 64 + iy) * 64 + ix) * 64;
      const unsigned dlds = lds_addr32(dst);
      #pragma unroll
      for (int j = 0; j < 8; ++j) {
        async_lds_b128_gv(dlds + (unsigned)(j * 16),
                          (unsigned long long)(uintptr_t)(src + j * 8));
      }
      const float r0 = (cy - lin64(iy)) * 64.0f;
      const float r1 = (cx - lin64(ix)) * 64.0f;
      dst[88] = (_Float16)r0; dst[89] = (_Float16)r1;
      dst[92] = (_Float16)0.f; dst[93] = (_Float16)0.f;
      dst[94] = (_Float16)0.f; dst[95] = (_Float16)0.f;
      areaSh[row] = fabsf(r0 * r1) + 1e-9f;
    } else {
      // shift-independent: 12-level hash encode + rel_cell
      const float x0 = (cy + 1.0f) * 0.5f;
      const float x1 = (cx + 1.0f) * 0.5f;
      for (int lvl = 0; lvl < 12; ++lvl) {
        const float resm1 = (float)((16 << lvl) - 1);
        const float gx = x1 * resm1, gy = x0 * resm1;
        float fx0 = fminf(fmaxf(floorf(gx), 0.0f), resm1);
        float fy0 = fminf(fmaxf(floorf(gy), 0.0f), resm1);
        const float fx = gx - fx0, fy = gy - fy0;
        const unsigned ix0 = (unsigned)fx0, iy0 = (unsigned)fy0;
        const unsigned ix1 = ix0 + 1u, iy1 = iy0 + 1u;
        const float2* tb = (const float2*)(tables + (long)lvl * 65536);
        const float2 e00 = tb[hash2(ix0, iy0)];
        const float2 e10 = tb[hash2(ix1, iy0)];
        const float2 e01 = tb[hash2(ix0, iy1)];
        const float2 e11 = tb[hash2(ix1, iy1)];
        const float w00 = (1.f - fx) * (1.f - fy), w10 = fx * (1.f - fy);
        const float w01 = (1.f - fx) * fy,         w11 = fx * fy;
        dst[64 + lvl * 2]     = (_Float16)(e00.x * w00 + e10.x * w10 + e01.x * w01 + e11.x * w11);
        dst[64 + lvl * 2 + 1] = (_Float16)(e00.y * w00 + e10.y * w10 + e01.y * w01 + e11.y * w11);
      }
      dst[90] = (_Float16)(cell[g * 2 + 0] * 64.0f);
      dst[91] = (_Float16)(cell[g * 2 + 1] * 64.0f);
    }
  }
  wait_asynccnt0();   // q_feat DMA landed in Xsh before the publishing barrier

  const _Float16* Wg[4] = { W0t, Wt, Wt + 65536, Wt + 2 * 65536 };
  const float*    Bg[4] = { b0, b1, b2, b3 };
  const int       Kl[4] = { 96, 256, 256, 256 };

  const int wv   = t >> 5;
  const int lane = t & 31;
  const int row0 = wv * 16;       // wave owns one 16-row block, all 256 cols
  _Float16* Xc = Xsh;
  _Float16* Yc = Ysh;

  for (int l = 0; l < 4; ++l) {
    __syncthreads();  // X ready (l==0) / prior layer done reading Wsh
    {
      // Stage this layer's weights into LDS via async DMA: row n = t,
      // 12..32 x 16B per thread, ASYNCcnt-tracked (no VGPR round trip).
      const int K = Kl[l];
      const unsigned dlds  = lds_addr32(Wsh + (size_t)t * WSR);
      const unsigned gbase = (unsigned)((size_t)t * K * 2);   // bytes
      const void* sbase = (const void*)Wg[l];
      const int nq = K >> 3;                                  // 16B chunks
      for (int j = 0; j < nq; ++j) {
        async_lds_b128_gvs(dlds + (unsigned)(j * 16),
                           gbase + (unsigned)(j * 16), sbase);
      }
      biasSh[t] = Bg[l][t];
      if (l < 3) __builtin_prefetch(Wg[l + 1] + (long)t * 256, 0, 0);
      wait_asynccnt0();
    }
    __syncthreads();

    const int nkb = Kl[l] >> 5;
    v8f zero = {0.f, 0.f, 0.f, 0.f, 0.f, 0.f, 0.f, 0.f};
    v8f acc[16];
    #pragma unroll
    for (int nb = 0; nb < 16; ++nb) acc[nb] = zero;

    for (int kb = 0; kb < nkb; ++kb) {
      const int k0 = kb * 32;
      const v16h a = ld_frag(Xc, XS, row0, k0, lane);
      #pragma unroll
      for (int nb = 0; nb < 16; ++nb) {
        const v16h bm = ld_frag(Wsh, WSR, nb * 16, k0, lane);
        acc[nb] = __builtin_amdgcn_wmma_f32_16x16x32_f16(
            false, a, false, bm, (short)0, acc[nb], false, false);
      }
    }

    // epilogue: bias + exact GELU -> f16 ping-pong buffer
    const int nlo = lane & 15;
    const int hi  = lane >> 4;
    #pragma unroll
    for (int nb = 0; nb < 16; ++nb) {
      const int n  = nb * 16 + nlo;
      const float bb = biasSh[n];
      const int mb = row0 + hi * 8;
      #pragma unroll
      for (int j = 0; j < 8; ++j) {
        Yc[(mb + j) * XS + n] = (_Float16)gelu_exact(acc[nb][j] + bb);
      }
    }
    _Float16* tmp = Xc; Xc = Yc; Yc = tmp;
  }
  __syncthreads();

  // -------- final layer 256 -> 3 (VALU, f32) --------
  if (t < MROWS) {
    const _Float16* xr = Xc + t * XS;
    float a0 = b4[0], a1 = b4[1], a2 = b4[2];
    for (int k = 0; k < 256; ++k) {
      const float xv = (float)xr[k];
      a0 = fmaf(xv, W4t[k],       a0);
      a1 = fmaf(xv, W4t[256 + k], a1);
      a2 = fmaf(xv, W4t[512 + k], a2);
    }
    predSh[t * 3 + 0] = a0; predSh[t * 3 + 1] = a1; predSh[t * 3 + 2] = a2;
  }
  __syncthreads();

  // -------- area blend (areas list reversed in reference) --------
  if (t < PTS_PER_WG) {
    const int p = t;
    const long g = (long)blockIdx.x * PTS_PER_WG + p;
    const float A0 = areaSh[p],      A1 = areaSh[32 + p];
    const float A2 = areaSh[64 + p], A3 = areaSh[96 + p];
    const float inv = 1.0f / (A0 + A1 + A2 + A3);
    const float w0 = A3 * inv, w1 = A2 * inv, w2 = A1 * inv, w3 = A0 * inv;
    #pragma unroll
    for (int o = 0; o < 3; ++o) {
      out[g * 3 + o] = predSh[(p) * 3 + o]       * w0
                     + predSh[(32 + p) * 3 + o]  * w1
                     + predSh[(64 + p) * 3 + o]  * w2
                     + predSh[(96 + p) * 3 + o]  * w3;
    }
  }
}

// ------------------------- launcher ----------------------------------------

extern "C" void kernel_launch(void* const* d_in, const int* in_sizes, int n_in,
                              void* d_out, int out_size, void* d_ws, size_t ws_size,
                              hipStream_t stream) {
  (void)in_sizes; (void)n_in; (void)out_size; (void)ws_size;
  const float* feat   = (const float*)d_in[0];
  const float* coord  = (const float*)d_in[1];
  const float* cell   = (const float*)d_in[2];
  const float* tables = (const float*)d_in[3];
  const float* W0 = (const float*)d_in[4];  const float* b0 = (const float*)d_in[5];
  const float* W1 = (const float*)d_in[6];  const float* b1 = (const float*)d_in[7];
  const float* W2 = (const float*)d_in[8];  const float* b2 = (const float*)d_in[9];
  const float* W3 = (const float*)d_in[10]; const float* b3 = (const float*)d_in[11];
  const float* W4 = (const float*)d_in[12]; const float* b4 = (const float*)d_in[13];
  float* out = (float*)d_out;

  char* ws = (char*)d_ws;
  _Float16* featT = (_Float16*)ws;                   // 2,097,152 B
  _Float16* W0t   = (_Float16*)(ws + 2097152);       //    49,152 B
  _Float16* Wt    = (_Float16*)(ws + 2146304);       //   393,216 B
  float*    W4t   = (float*)   (ws + 2539520);       //     3,072 B

  hipLaunchKernelGGL(prep_feat,    dim3(4096), dim3(256), 0, stream, feat, featT);
  hipLaunchKernelGGL(prep_weights, dim3(868),  dim3(256), 0, stream,
                     W0, W1, W2, W3, W4, W0t, Wt, W4t);
  hipLaunchKernelGGL(liif_fused,   dim3(8192), dim3(256), 0, stream,
                     featT, coord, cell, tables, W0t, Wt, W4t,
                     b0, b1, b2, b3, b4, out);
}
